// EnhancedDecoder_8598524526857
// MI455X (gfx1250) — matrix operand order repaired and checked by reading, not compiled, not run on hardware
//
#include <hip/hip_runtime.h>
#include <math.h>

// ---------------------------------------------------------------------------
// Types / helpers
// ---------------------------------------------------------------------------
typedef __attribute__((ext_vector_type(16))) unsigned short us16;
typedef __attribute__((ext_vector_type(16))) __bf16 v16bf;
typedef __attribute__((ext_vector_type(8)))  float  v8f;

constexpr int TM = 64;   // Cout tile per workgroup
constexpr int TN = 64;   // pixel tile per workgroup
constexpr int TK = 64;   // K per staging round (2 WMMA K-steps)

__device__ __forceinline__ unsigned short f2bf(float f) {
  unsigned u = __builtin_bit_cast(unsigned, f);
  unsigned r = u + 0x7FFFu + ((u >> 16) & 1u);   // round-to-nearest-even
  return (unsigned short)(r >> 16);
}
__device__ __forceinline__ float gelu_f(float x) {
  return 0.5f * x * (1.0f + erff(x * 0.70710678118654752f));
}
__device__ __forceinline__ float sigmoid_f(float x) {
  return 1.0f / (1.0f + __expf(-x));
}

// ---------------------------------------------------------------------------
// Implicit-GEMM conv (stride 1), per-sample (wBatched=1) or shared weights.
//   X  : bf16 activations (B, Cin, H, W); W = 1<<wshift
//   Wt : bf16 weights (wBatched? B:1, Cout, K), K = Cin*KKT, K % 64 == 0
//   out = act( BN( fa * (Wt @ im2col(X)) ) ) -> f32 and/or bf16 channel slice
// 8 wave32s: 4(M) x 2(N); per staging round each wave issues 4
// v_wmma_f32_16x16x32_bf16 (2 K-substeps x 2 N micro-tiles).
// A tile row-major [64][64]; B tile pixel-major [64][64] so every fragment is
// contiguous 16B/32B LDS reads (ds_load_b128) instead of scalar gathers.
// ---------------------------------------------------------------------------
template<int KKT>
__global__ __launch_bounds__(256) void wmma_conv_kernel(
    const unsigned short* __restrict__ X,
    const unsigned short* __restrict__ Wt, int wBatched,
    int Cin, int H, int Wd, int wshift, int Cout, int pad,
    const float* __restrict__ fa,
    const float* __restrict__ bng, const float* __restrict__ bnb,
    const float* __restrict__ bnm, const float* __restrict__ bnv,
    int act_gelu,
    float* __restrict__ outF,
    unsigned short* __restrict__ outBF, int bfC, int bfCoff)
{
  constexpr int KW = (KKT == 9) ? 3 : 1;
  const int HW = H * Wd;
  const int K  = Cin * KKT;
  const int wmask = Wd - 1;
  const int b       = blockIdx.z;
  const int coBase  = blockIdx.y * TM;
  const int pixBase = blockIdx.x * TN;

  __shared__ __align__(64) unsigned short As[TM][TK];  // weights (M x K)
  __shared__ __align__(64) unsigned short Bs[TN][TK];  // im2col, pixel-major

  const int tid  = threadIdx.x;
  const int lane = tid & 31;
  const int wid  = tid >> 5;
  const int m_off = (wid >> 1) * 16;          // 4 waves across M
  const int n_off = (wid & 1) * 32;           // 2 waves across N, 2 tiles each
  const int hi    = lane >> 4;
  const int l15   = lane & 15;

  v8f acc0 = {};
  v8f acc1 = {};

  const size_t wbase = (size_t)(wBatched ? b : 0) * (size_t)Cout * K;
  const size_t xbase = (size_t)b * (size_t)Cin * HW;

  // A staging: thread -> (row 0..63, 16 contiguous K)   = 2 x b128
  const int arow_t = tid >> 2;
  const int akc_t  = (tid & 3) * 16;
  // B staging: thread -> (K row 0..63, 16 contiguous pixels)
  const int bkl = tid >> 2;
  const int bn0 = (tid & 3) * 16;

  for (int k0 = 0; k0 < K; k0 += TK) {
    // ---- stage A tile ----
    {
      uint4 a0 = {0u, 0u, 0u, 0u}, a1 = {0u, 0u, 0u, 0u};
      int aco = coBase + arow_t;
      if (aco < Cout) {
        const unsigned short* wp = Wt + wbase + (size_t)aco * K + k0 + akc_t;
        a0 = *(const uint4*)wp;
        a1 = *(const uint4*)(wp + 8);
        if (k0 + TK < K) __builtin_prefetch(wp + TK, 0, 1);  // global_prefetch_b8
      }
      *(uint4*)(&As[arow_t][akc_t])     = a0;
      *(uint4*)(&As[arow_t][akc_t + 8]) = a1;
    }
    // ---- stage B tile (im2col with zero padding), transposed into LDS ----
    {
      int k  = k0 + bkl;
      int ci, ky, kx;
      if (KKT == 1) { ci = k; ky = 0; kx = 0; }
      else {
        ci = k / 9;
        int rr = k - ci * 9;
        ky = rr / 3;
        kx = rr - ky * 3;
      }
      const unsigned short* xc = X + xbase + (size_t)ci * HW;
      #pragma unroll
      for (int j = 0; j < 16; ++j) {
        int p = pixBase + bn0 + j;
        int y = (p >> wshift) + ky - pad;
        int x = (p & wmask) + kx - pad;
        unsigned short val = 0;
        if ((unsigned)y < (unsigned)H && (unsigned)x < (unsigned)Wd)
          val = xc[y * Wd + x];
        Bs[bn0 + j][bkl] = val;
      }
    }
    __syncthreads();

    // ---- 2 K-substeps x 2 N micro-tiles: 4 WMMAs per wave ----
    const int arow = m_off + l15;
    #pragma unroll
    for (int ks = 0; ks < 2; ++ks) {
      // A fragment: K = ks*32 + hi*16 .. +15 contiguous (matches B)
      union { us16 v; uint4 q[2]; } af;
      af.q[0] = *(const uint4*)(&As[arow][ks * 32 + hi * 16]);
      af.q[1] = *(const uint4*)(&As[arow][ks * 32 + hi * 16 + 8]);
      // B fragments: lane holds col N=l15, K = ks*32 + hi*16 .. +15 contiguous
      const us16 bf0 = *(const us16*)(&Bs[n_off + l15][ks * 32 + hi * 16]);
      const us16 bf1 = *(const us16*)(&Bs[n_off + 16 + l15][ks * 32 + hi * 16]);
      acc0 = __builtin_amdgcn_wmma_f32_16x16x32_bf16(
          false, __builtin_bit_cast(v16bf, af.v),
          false, __builtin_bit_cast(v16bf, bf0),
          (short)0, acc0, false, false);
      acc1 = __builtin_amdgcn_wmma_f32_16x16x32_bf16(
          false, __builtin_bit_cast(v16bf, af.v),
          false, __builtin_bit_cast(v16bf, bf1),
          (short)0, acc1, false, false);
    }
    __syncthreads();
  }

  // ---- epilogue: fa -> BN -> GELU -> f32 / bf16 stores ----
  // D layout: lane L holds N = L%16; VGPR v holds M = v + (L<16?0:8)
  const int mb = hi * 8;
  #pragma unroll
  for (int j = 0; j < 2; ++j) {
    const v8f& acc = j ? acc1 : acc0;
    const int p = pixBase + n_off + j * 16 + l15;
    #pragma unroll
    for (int v = 0; v < 8; ++v) {
      int co = coBase + m_off + mb + v;
      if (co < Cout) {
        float val = acc[v];
        if (fa)  val *= fa[b * Cout + co];
        if (bng) {
          float sc = bng[co] * rsqrtf(bnv[co] + 1e-5f);
          val = (val - bnm[co]) * sc + bnb[co];
        }
        if (act_gelu) val = gelu_f(val);
        if (outF)  outF[((size_t)b * Cout + co) * HW + p] = val;
        if (outBF) outBF[((size_t)b * bfC + bfCoff + co) * HW + p] = f2bf(val);
      }
    }
  }
}

// ---------------------------------------------------------------------------
// Small / glue kernels
// ---------------------------------------------------------------------------
__global__ __launch_bounds__(256) void gap_kernel(const float* __restrict__ x,
                                                  float* __restrict__ g, int HW) {
  __shared__ float sm[256];
  const float* p = x + (size_t)blockIdx.x * HW;
  float s = 0.f;
  for (int i = threadIdx.x; i < HW; i += 256) s += p[i];
  sm[threadIdx.x] = s;
  __syncthreads();
  for (int off = 128; off > 0; off >>= 1) {
    if (threadIdx.x < off) sm[threadIdx.x] += sm[threadIdx.x + off];
    __syncthreads();
  }
  if (threadIdx.x == 0) g[blockIdx.x] = sm[0] / (float)HW;
}

// mode: 0=+bias, 1=BN+ReLU, 2=+bias,sigmoid, 3=+bias,GELU
__global__ __launch_bounds__(256) void matvec_post_kernel(
    const float* __restrict__ W, const float* __restrict__ x,
    const float* __restrict__ bias,
    const float* __restrict__ bg, const float* __restrict__ bb,
    const float* __restrict__ bm, const float* __restrict__ bv,
    float* __restrict__ y, int Bn, int O, int I, int mode)
{
  int idx = blockIdx.x * 256 + threadIdx.x;
  if (idx >= Bn * O) return;
  int b = idx / O, o = idx - b * O;
  const float* wr = W + (size_t)o * I;
  const float* xr = x + (size_t)b * I;
  float s = 0.f;
  for (int i = 0; i < I; ++i) s += wr[i] * xr[i];
  if (mode == 1) {
    float sc = bg[o] * rsqrtf(bv[o] + 1e-5f);
    s = (s - bm[o]) * sc + bb[o];
    s = fmaxf(s, 0.f);
  } else {
    if (bias) s += bias[o];
    if (mode == 2) s = sigmoid_f(s);
    else if (mode == 3) s = gelu_f(s);
  }
  y[idx] = s;
}

__global__ void softmax_rows_kernel(float* y, int Bn, int n) {
  int b = blockIdx.x * blockDim.x + threadIdx.x;
  if (b >= Bn) return;
  float* r = y + (size_t)b * n;
  float m = r[0];
  for (int i = 1; i < n; ++i) m = fmaxf(m, r[i]);
  float s = 0.f;
  for (int i = 0; i < n; ++i) { r[i] = __expf(r[i] - m); s += r[i]; }
  float inv = 1.f / s;
  for (int i = 0; i < n; ++i) r[i] *= inv;
}

// aw[b][co][ci*KK+kk] = ca[b][ci] * sa[b][kk] * sum_q ka[b][q] * w[q][co][ci][kk]
__global__ __launch_bounds__(256) void weight_mix_kernel(
    const float* __restrict__ w, const float* __restrict__ ka,
    const float* __restrict__ sa, const float* __restrict__ ca,
    unsigned short* __restrict__ out,
    int kn, int Cout, int Cin, int KK, size_t tot)
{
  size_t idx = (size_t)blockIdx.x * 256 + threadIdx.x;
  if (idx >= tot) return;
  int kk = (int)(idx % KK);
  size_t t = idx / KK;
  int ci = (int)(t % Cin); t /= Cin;
  int co = (int)(t % Cout);
  int b  = (int)(t / Cout);
  size_t stride = (size_t)Cout * Cin * KK;
  size_t off = ((size_t)co * Cin + ci) * KK + kk;
  float s = 0.f;
  for (int q = 0; q < kn; ++q) s += ka[b * kn + q] * w[q * stride + off];
  s *= ca[b * Cin + ci];
  if (sa) s *= sa[b * KK + kk];
  out[idx] = f2bf(s);
}

__global__ __launch_bounds__(256) void f32_to_bf16_kernel(
    const float* __restrict__ x, unsigned short* __restrict__ y, size_t n) {
  size_t i = (size_t)blockIdx.x * 256 + threadIdx.x;
  if (i < n) y[i] = f2bf(x[i]);
}

// bilinear 2x, align_corners=True; writes f32 and optional bf16
__global__ __launch_bounds__(256) void upsample2x_kernel(
    const float* __restrict__ x, float* __restrict__ oF,
    unsigned short* __restrict__ oBF, int H, int W, size_t tot)
{
  int Ho = 2 * H, Wo = 2 * W;
  size_t idx = (size_t)blockIdx.x * 256 + threadIdx.x;
  if (idx >= tot) return;
  int xo = (int)(idx % Wo);
  size_t t = idx / Wo;
  int yo = (int)(t % Ho);
  t /= Ho;                                  // t = b*C + c
  float fy = (float)yo * (float)(H - 1) / (float)(Ho - 1);
  float fx = (float)xo * (float)(W - 1) / (float)(Wo - 1);
  int y0 = (int)fy; int y1 = y0 + 1; if (y1 > H - 1) y1 = H - 1;
  int x0 = (int)fx; int x1 = x0 + 1; if (x1 > W - 1) x1 = W - 1;
  float wy = fy - (float)y0, wx = fx - (float)x0;
  const float* p = x + t * (size_t)(H * W);
  float top = p[y0 * W + x0] * (1.f - wx) + p[y0 * W + x1] * wx;
  float bot = p[y1 * W + x0] * (1.f - wx) + p[y1 * W + x1] * wx;
  float v = top * (1.f - wy) + bot * wy;
  oF[idx] = v;
  if (oBF) oBF[idx] = f2bf(v);
}

__global__ __launch_bounds__(256) void add_kernel(const float* __restrict__ a,
                                                  const float* __restrict__ b,
                                                  float* __restrict__ c, size_t n) {
  size_t i = (size_t)blockIdx.x * 256 + threadIdx.x;
  if (i < n) c[i] = a[i] + b[i];
}

__global__ __launch_bounds__(256) void scale_ch_kernel(
    const float* __restrict__ x, const float* __restrict__ a,
    float* __restrict__ oF, unsigned short* __restrict__ oBF, int HW, size_t n) {
  size_t i = (size_t)blockIdx.x * 256 + threadIdx.x;
  if (i >= n) return;
  float v = x[i] * a[i / HW];   // a indexed by (b*C + c)
  oF[i] = v;
  if (oBF) oBF[i] = f2bf(v);
}

__global__ __launch_bounds__(256) void final_conv_kernel(
    const float* __restrict__ x, const float* __restrict__ w,
    const float* __restrict__ bias, float* __restrict__ out,
    int Bn, int Cin, int H, int Wd)
{
  int HW = H * Wd;
  int idx = blockIdx.x * 256 + threadIdx.x;
  if (idx >= Bn * HW) return;
  int b = idx / HW, p = idx - b * HW;
  int py = p / Wd, px = p - py * Wd;
  float s = bias[0];
  for (int ci = 0; ci < Cin; ++ci) {
    const float* xc = x + ((size_t)b * Cin + ci) * HW;
    const float* wc = w + ci * 9;
    for (int ky = 0; ky < 3; ++ky) {
      int y = py + ky - 1;
      if ((unsigned)y >= (unsigned)H) continue;
      for (int kx = 0; kx < 3; ++kx) {
        int xx = px + kx - 1;
        if ((unsigned)xx >= (unsigned)Wd) continue;
        s += wc[ky * 3 + kx] * xc[y * Wd + xx];
      }
    }
  }
  out[idx] = sigmoid_f(s);
}

// ---------------------------------------------------------------------------
// Host-side orchestration
// ---------------------------------------------------------------------------
struct Arena {
  char* base; size_t off; size_t cap;
  void* alloc(size_t bytes) {
    size_t a = (off + 255) & ~(size_t)255;
    off = a + bytes;
    return base + a;
  }
};
struct BnP { const float *g, *b, *m, *v; };
struct OdP {
  const float *w, *fcw; BnP fcbn;
  const float *chw, *chb, *ftw, *ftb, *spw, *spb, *knw, *knb;
  int kn, att;
};
struct CaP { const float *w1, *b1, *w2, *b2; };

static inline int log2i(int v) { int s = 0; while ((1 << s) < v) ++s; return s; }

static void conv_launch(hipStream_t st, const unsigned short* xBF,
                        const unsigned short* wBF, int wBatched,
                        int B, int Cin, int H, int W, int Cout, int ks,
                        const float* fa, const BnP* bn, int gelu,
                        float* outF, unsigned short* outBF, int bfC, int bfCoff)
{
  const int HW = H * W;
  dim3 grid(HW / TN, (Cout + TM - 1) / TM, B);
  if (ks == 3)
    wmma_conv_kernel<9><<<grid, 256, 0, st>>>(
        xBF, wBF, wBatched, Cin, H, W, log2i(W), Cout, 1,
        fa, bn ? bn->g : nullptr, bn ? bn->b : nullptr,
        bn ? bn->m : nullptr, bn ? bn->v : nullptr,
        gelu, outF, outBF, bfC, bfCoff);
  else
    wmma_conv_kernel<1><<<grid, 256, 0, st>>>(
        xBF, wBF, wBatched, Cin, H, W, log2i(W), Cout, 0,
        fa, bn ? bn->g : nullptr, bn ? bn->b : nullptr,
        bn ? bn->m : nullptr, bn ? bn->v : nullptr,
        gelu, outF, outBF, bfC, bfCoff);
}

static void odconv_run(hipStream_t st, Arena s /*scratch by value*/,
                       const float* xF, const unsigned short* xBF, const OdP& od,
                       int B, int Cin, int H, int W, int Cout, int ks,
                       const BnP* bn, int gelu,
                       float* outF, unsigned short* outBF, int bfC, int bfCoff)
{
  const int HW = H * W, KK = ks * ks, att = od.att;
  float* g   = (float*)s.alloc(sizeof(float) * (size_t)B * Cin);
  float* g2  = (float*)s.alloc(sizeof(float) * (size_t)B * att);
  float* ca  = (float*)s.alloc(sizeof(float) * (size_t)B * Cin);
  float* fa  = (float*)s.alloc(sizeof(float) * (size_t)B * Cout);
  float* sa  = (ks > 1) ? (float*)s.alloc(sizeof(float) * (size_t)B * KK) : nullptr;
  float* kar = (float*)s.alloc(sizeof(float) * (size_t)B * od.kn);
  size_t totW = (size_t)B * Cout * Cin * KK;
  unsigned short* aw = (unsigned short*)s.alloc(totW * 2);

  gap_kernel<<<B * Cin, 256, 0, st>>>(xF, g, HW);
  int t;
  t = B * att;
  matvec_post_kernel<<<(t + 255) / 256, 256, 0, st>>>(
      od.fcw, g, nullptr, od.fcbn.g, od.fcbn.b, od.fcbn.m, od.fcbn.v,
      g2, B, att, Cin, 1);
  t = B * Cin;
  matvec_post_kernel<<<(t + 255) / 256, 256, 0, st>>>(
      od.chw, g2, od.chb, nullptr, nullptr, nullptr, nullptr, ca, B, Cin, att, 2);
  t = B * Cout;
  matvec_post_kernel<<<(t + 255) / 256, 256, 0, st>>>(
      od.ftw, g2, od.ftb, nullptr, nullptr, nullptr, nullptr, fa, B, Cout, att, 2);
  if (sa) {
    t = B * KK;
    matvec_post_kernel<<<(t + 255) / 256, 256, 0, st>>>(
        od.spw, g2, od.spb, nullptr, nullptr, nullptr, nullptr, sa, B, KK, att, 2);
  }
  t = B * od.kn;
  matvec_post_kernel<<<(t + 255) / 256, 256, 0, st>>>(
      od.knw, g2, od.knb, nullptr, nullptr, nullptr, nullptr, kar, B, od.kn, att, 0);
  softmax_rows_kernel<<<1, 32, 0, st>>>(kar, B, od.kn);

  weight_mix_kernel<<<(unsigned)((totW + 255) / 256), 256, 0, st>>>(
      od.w, kar, sa, ca, aw, od.kn, Cout, Cin, KK, totW);

  conv_launch(st, xBF, aw, 1, B, Cin, H, W, Cout, ks, fa, bn, gelu,
              outF, outBF, bfC, bfCoff);
}

static void channel_attn_run(hipStream_t st, Arena s, const CaP& ca, const float* tF,
                             int B, int C, int HW, float* outF, unsigned short* outBF)
{
  int h = C / 8;
  float* g  = (float*)s.alloc(sizeof(float) * (size_t)B * C);
  float* g2 = (float*)s.alloc(sizeof(float) * (size_t)B * h);
  float* a  = (float*)s.alloc(sizeof(float) * (size_t)B * C);
  gap_kernel<<<B * C, 256, 0, st>>>(tF, g, HW);
  int t = B * h;
  matvec_post_kernel<<<(t + 255) / 256, 256, 0, st>>>(
      ca.w1, g, ca.b1, nullptr, nullptr, nullptr, nullptr, g2, B, h, C, 3);
  t = B * C;
  matvec_post_kernel<<<(t + 255) / 256, 256, 0, st>>>(
      ca.w2, g2, ca.b2, nullptr, nullptr, nullptr, nullptr, a, B, C, h, 2);
  size_t n = (size_t)B * C * HW;
  scale_ch_kernel<<<(unsigned)((n + 255) / 256), 256, 0, st>>>(tF, a, outF, outBF, HW, n);
}

// ---------------------------------------------------------------------------
extern "C" void kernel_launch(void* const* d_in, const int* in_sizes, int n_in,
                              void* d_out, int out_size, void* d_ws, size_t ws_size,
                              hipStream_t stream) {
  (void)in_sizes; (void)n_in; (void)out_size;
  const int B = 8;
  int ii = 0;
  auto nf = [&]() -> const float* { return (const float*)d_in[ii++]; };
  auto loadBn = [&](BnP& bn) { bn.g = nf(); bn.b = nf(); bn.m = nf(); bn.v = nf(); };
  auto loadOd = [&](OdP& o, int in_c, int ks, int kn) {
    o.w = nf(); o.fcw = nf(); loadBn(o.fcbn);
    o.chw = nf(); o.chb = nf(); o.ftw = nf(); o.ftb = nf();
    if (ks > 1) { o.spw = nf(); o.spb = nf(); } else { o.spw = o.spb = nullptr; }
    if (kn > 1) { o.knw = nf(); o.knb = nf(); } else { o.knw = o.knb = nullptr; }
    o.kn = kn;
    int att = in_c / 16; if (att < 16) att = 16;
    o.att = att;
  };
  auto loadCa = [&](CaP& c) { c.w1 = nf(); c.b1 = nf(); c.w2 = nf(); c.b2 = nf(); };

  const float* f[4];
  for (int i = 0; i < 4; ++i) f[i] = nf();
  OdP adOd[4]; BnP adBn[4]; OdP adOd2{};
  int inC[4] = {256, 512, 1024, 2048};
  for (int i = 0; i < 4; ++i) {
    loadOd(adOd[i], inC[i], 3, 2);
    loadBn(adBn[i]);
    if (i == 3) loadOd(adOd2, 256, 3, 2);
  }
  const float* fusion_w = nf();
  OdP r1od1, r1od2; BnP r1bn1, r1bn2; CaP r1ca;
  loadOd(r1od1, 256, 3, 4); loadBn(r1bn1);
  loadOd(r1od2, 512, 3, 4); loadBn(r1bn2);
  loadCa(r1ca);
  OdP u1od; BnP u1bn; loadOd(u1od, 256, 3, 2); loadBn(u1bn);
  OdP r2od1, r2od2, r2sc; BnP r2bn1, r2bn2; CaP r2ca;
  loadOd(r2od1, 128, 3, 2); loadBn(r2bn1);
  loadOd(r2od2, 256, 3, 2); loadBn(r2bn2);
  loadOd(r2sc, 128, 1, 4);
  loadCa(r2ca);
  OdP u2od; BnP u2bn; loadOd(u2od, 64, 3, 2); loadBn(u2bn);
  const float* final_w = nf();
  const float* final_b = nf();

  Arena A{(char*)d_ws, 0, ws_size};

  // ---- Stage A: adapters -> concat bf16 (B, 1024, 64, 64) ----
  unsigned short* concatBF = (unsigned short*)A.alloc((size_t)B * 1024 * 4096 * 2);
  for (int i = 0; i < 3; ++i) {
    Arena s = A;
    size_t nx = (size_t)B * inC[i] * 4096;
    unsigned short* xbf = (unsigned short*)s.alloc(nx * 2);
    f32_to_bf16_kernel<<<(unsigned)((nx + 255) / 256), 256, 0, stream>>>(f[i], xbf, nx);
    odconv_run(stream, s, f[i], xbf, adOd[i], B, inC[i], 64, 64, 256, 3,
               &adBn[i], 1, nullptr, concatBF, 1024, 256 * i);
  }
  {
    Arena s = A;
    size_t nx = (size_t)B * 2048 * 1024;
    unsigned short* xbf = (unsigned short*)s.alloc(nx * 2);
    f32_to_bf16_kernel<<<(unsigned)((nx + 255) / 256), 256, 0, stream>>>(f[3], xbf, nx);
    float* h3F = (float*)s.alloc((size_t)B * 256 * 1024 * 4);
    odconv_run(stream, s, f[3], xbf, adOd[3], B, 2048, 32, 32, 256, 3,
               &adBn[3], 1, h3F, nullptr, 0, 0);
    size_t nu = (size_t)B * 256 * 4096;
    float* h3uF = (float*)s.alloc(nu * 4);
    unsigned short* h3uBF = (unsigned short*)s.alloc(nu * 2);
    upsample2x_kernel<<<(unsigned)((nu + 255) / 256), 256, 0, stream>>>(h3F, h3uF, h3uBF, 32, 32, nu);
    odconv_run(stream, s, h3uF, h3uBF, adOd2, B, 256, 64, 64, 256, 3,
               nullptr, 0, nullptr, concatBF, 1024, 768);
  }

  // ---- Stage B: fusion 1x1 conv (shared weights) ----
  float* fusedF = (float*)A.alloc((size_t)B * 256 * 4096 * 4);
  unsigned short* fusedBF = (unsigned short*)A.alloc((size_t)B * 256 * 4096 * 2);
  {
    Arena s = A;
    size_t nw = (size_t)256 * 1024;
    unsigned short* wbf = (unsigned short*)s.alloc(nw * 2);
    f32_to_bf16_kernel<<<(unsigned)((nw + 255) / 256), 256, 0, stream>>>(fusion_w, wbf, nw);
    conv_launch(stream, concatBF, wbf, 0, B, 1024, 64, 64, 256, 1,
                nullptr, nullptr, 0, fusedF, fusedBF, 256, 0);
  }

  // ---- Stage C: res1 ----
  float* r1F = (float*)A.alloc((size_t)B * 256 * 4096 * 4);
  unsigned short* r1BF = (unsigned short*)A.alloc((size_t)B * 256 * 4096 * 2);
  {
    Arena s = A;
    float* hF = (float*)s.alloc((size_t)B * 512 * 4096 * 4);
    unsigned short* hBF = (unsigned short*)s.alloc((size_t)B * 512 * 4096 * 2);
    odconv_run(stream, s, fusedF, fusedBF, r1od1, B, 256, 64, 64, 512, 3,
               &r1bn1, 1, hF, hBF, 512, 0);
    float* h2F = (float*)s.alloc((size_t)B * 256 * 4096 * 4);
    odconv_run(stream, s, hF, hBF, r1od2, B, 512, 64, 64, 256, 3,
               &r1bn2, 0, h2F, nullptr, 0, 0);
    float* tF = (float*)s.alloc((size_t)B * 256 * 4096 * 4);
    size_t nt = (size_t)B * 256 * 4096;
    add_kernel<<<(unsigned)((nt + 255) / 256), 256, 0, stream>>>(h2F, fusedF, tF, nt);
    channel_attn_run(stream, s, r1ca, tF, B, 256, 4096, r1F, r1BF);
  }

  // ---- Stage D: up1 ----
  float* u1F = (float*)A.alloc((size_t)B * 128 * 16384 * 4);
  unsigned short* u1BF = (unsigned short*)A.alloc((size_t)B * 128 * 16384 * 2);
  {
    Arena s = A;
    size_t nu = (size_t)B * 256 * 16384;
    float* upF = (float*)s.alloc(nu * 4);
    unsigned short* upBF = (unsigned short*)s.alloc(nu * 2);
    upsample2x_kernel<<<(unsigned)((nu + 255) / 256), 256, 0, stream>>>(r1F, upF, upBF, 64, 64, nu);
    odconv_run(stream, s, upF, upBF, u1od, B, 256, 128, 128, 128, 3,
               &u1bn, 1, u1F, u1BF, 128, 0);
  }

  // ---- Stage E: res2 ----
  float* r2F = (float*)A.alloc((size_t)B * 64 * 16384 * 4);
  unsigned short* r2BF = (unsigned short*)A.alloc((size_t)B * 64 * 16384 * 2);
  {
    Arena s = A;
    float* hF = (float*)s.alloc((size_t)B * 256 * 16384 * 4);
    unsigned short* hBF = (unsigned short*)s.alloc((size_t)B * 256 * 16384 * 2);
    odconv_run(stream, s, u1F, u1BF, r2od1, B, 128, 128, 128, 256, 3,
               &r2bn1, 1, hF, hBF, 256, 0);
    float* h2F = (float*)s.alloc((size_t)B * 64 * 16384 * 4);
    odconv_run(stream, s, hF, hBF, r2od2, B, 256, 128, 128, 64, 3,
               &r2bn2, 0, h2F, nullptr, 0, 0);
    float* scF = (float*)s.alloc((size_t)B * 64 * 16384 * 4);
    odconv_run(stream, s, u1F, u1BF, r2sc, B, 128, 128, 128, 64, 1,
               nullptr, 0, scF, nullptr, 0, 0);
    float* tF = (float*)s.alloc((size_t)B * 64 * 16384 * 4);
    size_t nt = (size_t)B * 64 * 16384;
    add_kernel<<<(unsigned)((nt + 255) / 256), 256, 0, stream>>>(h2F, scF, tF, nt);
    channel_attn_run(stream, s, r2ca, tF, B, 64, 16384, r2F, r2BF);
  }

  // ---- Stage F: up2 ----
  float* u2F = (float*)A.alloc((size_t)B * 32 * 65536 * 4);
  {
    Arena s = A;
    size_t nu = (size_t)B * 64 * 65536;
    float* upF = (float*)s.alloc(nu * 4);
    unsigned short* upBF = (unsigned short*)s.alloc(nu * 2);
    upsample2x_kernel<<<(unsigned)((nu + 255) / 256), 256, 0, stream>>>(r2F, upF, upBF, 128, 128, nu);
    odconv_run(stream, s, upF, upBF, u2od, B, 64, 256, 256, 32, 3,
               &u2bn, 1, u2F, nullptr, 0, 0);
  }

  // ---- Stage G: final 3x3 conv (32->1) + sigmoid -> d_out ----
  {
    int nt = B * 65536;
    final_conv_kernel<<<(nt + 255) / 256, 256, 0, stream>>>(
        u2F, final_w, final_b, (float*)d_out, B, 32, 256, 256);
  }
}
